// HARTBinary_52467320487969
// MI455X (gfx1250) — matrix operand order repaired and checked by reading, not compiled, not run on hardware
//
#include <hip/hip_runtime.h>

// ---------------------------------------------------------------------------
// HART-style hypergraph transformer for MI455X (gfx1250), wave32 + WMMA.
// Compute-bound (~4 TFLOP GEMM vs ~0.5GB HBM): all GEMMs via
// v_wmma_f32_16x16x32_f16 (f16 in, f32 accum). One workgroup = one sequence,
// whole 2-layer encoder fused in LDS (~114KB of the 320KB/WGP).
// ---------------------------------------------------------------------------

typedef _Float16 v8h  __attribute__((ext_vector_type(8)));
typedef _Float16 v16h __attribute__((ext_vector_type(16)));
typedef float    v8f  __attribute__((ext_vector_type(8)));

#define EMBED   256
#define HEADS   4
#define DHEAD   64
#define HIDDEN  1024
#define NLAYER  2
#define NVOC    200000
#define NEDGE   16384
#define LSEQ    16
#define NNODE   16384
#define LHID    12
#define NBATCH  1024

#define S_EDGE  17           // 1 + LSEQ
#define S_NODE  13           // 1 + LHID
#define SROWS   17           // f32 row buffers sized for max S
#define MPAD    32           // M padded to 2 WMMA tiles
#define NWAVE   8            // 256 threads / 32

struct EncParams {
    const _Float16 *wqkv, *wout, *wff1, *wff2;   // f16, row-major [N][K]
    const float    *bqkv, *bout, *bff1, *bff2;
    const float    *ln1w, *ln1b, *ln2w, *ln2b;
};

struct EncSM {
    float     x[SROWS * EMBED];     // current activations (f32, residual)
    float     y[SROWS * EMBED];     // GEMM f32 output before residual+LN
    float     smask[32];            // attention key mask (1/0)
    _Float16  xh[MPAD * EMBED];     // f16 A-matrix (x or ctx), zero-padded rows
    _Float16  h16[MPAD * HIDDEN];   // f16 qkv output / FFN hidden
};

// ---- WMMA fragment loads --------------------------------------------------
// A: 16x32 f16, lane(l): row = l&15, K chunks at 8*(l>>4) and 16+8*(l>>4)
__device__ __forceinline__ v16h load_a_frag(const _Float16* A, int ldA,
                                            int m0, int ks, int lane) {
    const int l16 = lane & 15, half = lane >> 4;
    const _Float16* p = A + (m0 + l16) * ldA + ks * 32 + 8 * half;
    v8h c0 = *(const v8h*)p;
    v8h c1 = *(const v8h*)(p + 16);
    return __builtin_shufflevector(c0, c1, 0,1,2,3,4,5,6,7,8,9,10,11,12,13,14,15);
}
// B: 32x16 f16 of W^T from row-major W[N][K], K chunks at 8*(l>>4) and 16+8*(l>>4)
__device__ __forceinline__ v16h load_b_frag(const _Float16* W, int K,
                                            int n0, int ks, int lane) {
    const int l16 = lane & 15, half = lane >> 4;
    const _Float16* p = W + (size_t)(n0 + l16) * K + ks * 32 + 8 * half;
    v8h c0 = *(const v8h*)p;
    v8h c1 = *(const v8h*)(p + 16);
    return __builtin_shufflevector(c0, c1, 0,1,2,3,4,5,6,7,8,9,10,11,12,13,14,15);
}

// ---- Block GEMM: out[MPAD x N] = A[MPAD x K](f16) * W[N x K]^T + bias -----
template<bool OUT_F16, bool RELU>
__device__ void gemm_block(const _Float16* A, int ldA,
                           const _Float16* W, const float* bias, int N, int K,
                           _Float16* out16, float* out32, int ldO,
                           int wave, int lane) {
    const int l16 = lane & 15, half = lane >> 4;
    const int ntiles = N >> 4, ksteps = K >> 5;
    for (int nt = wave; nt < ntiles; nt += NWAVE) {
        const int n0 = nt * 16;
        const float bv = bias[n0 + l16];
        v8f acc0 = {bv,bv,bv,bv,bv,bv,bv,bv};
        v8f acc1 = acc0;
        for (int ks = 0; ks < ksteps; ++ks) {
            v16h b  = load_b_frag(W, K, n0, ks, lane);
            v16h a0 = load_a_frag(A, ldA, 0,  ks, lane);
            v16h a1 = load_a_frag(A, ldA, 16, ks, lane);
            acc0 = __builtin_amdgcn_wmma_f32_16x16x32_f16(false, a0, false, b,
                                                          (short)0, acc0, false, false);
            acc1 = __builtin_amdgcn_wmma_f32_16x16x32_f16(false, a1, false, b,
                                                          (short)0, acc1, false, false);
        }
        const int c = n0 + l16;
#pragma unroll
        for (int j = 0; j < 8; ++j) {
            int r0 = 8 * half + j, r1 = 16 + r0;
            float v0 = acc0[j], v1 = acc1[j];
            if (RELU) { v0 = fmaxf(v0, 0.f); v1 = fmaxf(v1, 0.f); }
            if (OUT_F16) {
                out16[r0 * ldO + c] = (_Float16)v0;
                out16[r1 * ldO + c] = (_Float16)v1;
            } else {
                if (r0 < SROWS) out32[r0 * ldO + c] = v0;
                if (r1 < SROWS) out32[r1 * ldO + c] = v1;
            }
        }
    }
}

// ---- residual + LayerNorm (f32, one wave per row) -------------------------
__device__ void resid_ln(float* x, const float* y, const float* w, const float* b,
                         int S, int wave, int lane) {
    for (int r = wave; r < S; r += NWAVE) {
        float vals[8], s = 0.f;
#pragma unroll
        for (int i = 0; i < 8; ++i) {
            float t = x[r * EMBED + lane + 32 * i] + y[r * EMBED + lane + 32 * i];
            vals[i] = t; s += t;
        }
        for (int off = 16; off; off >>= 1) s += __shfl_xor(s, off, 32);
        const float mu = s * (1.f / EMBED);
        float vs = 0.f;
#pragma unroll
        for (int i = 0; i < 8; ++i) { float d = vals[i] - mu; vs += d * d; }
        for (int off = 16; off; off >>= 1) vs += __shfl_xor(vs, off, 32);
        const float rinv = rsqrtf(vs * (1.f / EMBED) + 1e-5f);
#pragma unroll
        for (int i = 0; i < 8; ++i) {
            int c = lane + 32 * i;
            x[r * EMBED + c] = (vals[i] - mu) * rinv * w[c] + b[c];
        }
    }
}

// ---- attention: qkv(f16 in h16, ld=HIDDEN) -> ctx(f16 into xh) ------------
__device__ void attention(const _Float16* qkv, _Float16* ctx,
                          const float* smask, int S, int wave, int lane) {
    const float scale = 0.125f;               // 1/sqrt(64)
    const int ntask = HEADS * S;
    for (int t = wave; t < ntask; t += NWAVE) {
        const int h = t / S, q = t - h * S;
        const _Float16* qrow = qkv + q * HIDDEN + h * DHEAD;
        float sc = -1e9f;
        if (lane < S) {
            const _Float16* krow = qkv + lane * HIDDEN + EMBED + h * DHEAD;
            float acc = 0.f;
            for (int d = 0; d < DHEAD; ++d) acc += (float)qrow[d] * (float)krow[d];
            sc = (smask[lane] != 0.f) ? acc * scale : -1e9f;
        }
        float m = sc;
        for (int off = 16; off; off >>= 1) m = fmaxf(m, __shfl_xor(m, off, 32));
        float p = __expf(sc - m);
        float ps = p;
        for (int off = 16; off; off >>= 1) ps += __shfl_xor(ps, off, 32);
        const float a = p / ps;
#pragma unroll
        for (int hh = 0; hh < 2; ++hh) {
            const int d = hh * 32 + lane;
            float acc = 0.f;
            for (int j = 0; j < S; ++j) {
                float aj = __shfl(a, j, 32);
                acc += aj * (float)qkv[j * HIDDEN + 2 * EMBED + h * DHEAD + d];
            }
            ctx[q * EMBED + h * DHEAD + d] = (_Float16)acc;
        }
    }
}

// ---- fused 2-layer encoder over one sequence ------------------------------
__device__ void encoder(EncSM* sm, int S, const EncParams* P, int tid) {
    const int wave = tid >> 5, lane = tid & 31;
    for (int l = 0; l < NLAYER; ++l) {
        for (int i = tid; i < MPAD * EMBED; i += 256) {      // x -> xh (pad 0)
            int r = i >> 8, c = i & 255;
            sm->xh[i] = (r < S) ? (_Float16)sm->x[r * EMBED + c] : (_Float16)0.f;
        }
        __syncthreads();
        gemm_block<true, false>(sm->xh, EMBED, P->wqkv + (size_t)l * 3 * EMBED * EMBED,
                                P->bqkv + l * 3 * EMBED, 3 * EMBED, EMBED,
                                sm->h16, nullptr, HIDDEN, wave, lane);
        __syncthreads();
        attention(sm->h16, sm->xh, sm->smask, S, wave, lane);
        __syncthreads();
        gemm_block<false, false>(sm->xh, EMBED, P->wout + (size_t)l * EMBED * EMBED,
                                 P->bout + l * EMBED, EMBED, EMBED,
                                 nullptr, sm->y, EMBED, wave, lane);
        __syncthreads();
        resid_ln(sm->x, sm->y, P->ln1w + l * EMBED, P->ln1b + l * EMBED, S, wave, lane);
        __syncthreads();
        for (int i = tid; i < MPAD * EMBED; i += 256) {      // x -> xh again
            int r = i >> 8, c = i & 255;
            sm->xh[i] = (r < S) ? (_Float16)sm->x[r * EMBED + c] : (_Float16)0.f;
        }
        __syncthreads();
        gemm_block<true, true>(sm->xh, EMBED, P->wff1 + (size_t)l * HIDDEN * EMBED,
                               P->bff1 + l * HIDDEN, HIDDEN, EMBED,
                               sm->h16, nullptr, HIDDEN, wave, lane);
        __syncthreads();
        gemm_block<false, false>(sm->h16, HIDDEN, P->wff2 + (size_t)l * EMBED * HIDDEN,
                                 P->bff2 + l * EMBED, EMBED, HIDDEN,
                                 nullptr, sm->y, EMBED, wave, lane);
        __syncthreads();
        resid_ln(sm->x, sm->y, P->ln2w + l * EMBED, P->ln2b + l * EMBED, S, wave, lane);
        __syncthreads();
    }
}

// ---- edge-stage kernel: one block per hyperedge ---------------------------
__global__ __launch_bounds__(256)
void edge_encoder_kernel(const float* __restrict__ unity,
                         const float* __restrict__ pos_table,
                         const int* __restrict__ xid,
                         const unsigned char* __restrict__ xmask,
                         const int* __restrict__ xpos,
                         const float* __restrict__ ovr,
                         const int* __restrict__ inv, int useOvr,
                         EncParams prm, float* __restrict__ hedge) {
    extern __shared__ char smraw[];
    EncSM* sm = (EncSM*)smraw;
    const int n = blockIdx.x, tid = threadIdx.x;
    for (int i = tid; i < S_EDGE * EMBED; i += 256) {
        int r = i >> 8, c = i & 255;
        float v = 0.f;
        if (r > 0) {
            int id = xid[(size_t)n * LSEQ + r - 1];
            const float* base = unity + (size_t)id * EMBED;
            if (useOvr) { int ii = inv[id]; if (ii >= 0) base = ovr + (size_t)ii * EMBED; }
            v = base[c];
        }
        v += pos_table[(size_t)xpos[(size_t)n * (LSEQ + 1) + r] * EMBED + c];
        sm->x[i] = v;
    }
    if (tid < 32) {
        float mv = 0.f;
        if (tid == 0) mv = 1.f;
        else if (tid < S_EDGE) mv = xmask[(size_t)n * LSEQ + tid - 1] ? 1.f : 0.f;
        sm->smask[tid] = mv;
    }
    __syncthreads();
    encoder(sm, S_EDGE, &prm, tid);
    hedge[(size_t)n * EMBED + tid] = sm->x[tid];            // token 0
}

// ---- node-stage kernel: one block per subgraph node -----------------------
__global__ __launch_bounds__(256)
void node_encoder_kernel(const float* __restrict__ hedge,
                         const float* __restrict__ pad_emb,
                         const float* __restrict__ cls_emb,
                         const int* __restrict__ hid,
                         const unsigned char* __restrict__ hmask,
                         EncParams prm, float* __restrict__ ovr) {
    extern __shared__ char smraw[];
    EncSM* sm = (EncSM*)smraw;
    const int n = blockIdx.x, tid = threadIdx.x;
    for (int i = tid; i < S_NODE * EMBED; i += 256) {
        int r = i >> 8, c = i & 255;
        float v;
        if (r == 0) v = cls_emb[c];
        else {
            int id = hid[(size_t)n * LHID + r - 1];
            if (id < NEDGE)       v = hedge[(size_t)id * EMBED + c];
            else if (id == NEDGE) v = pad_emb[c];
            else                  v = cls_emb[c];
        }
        sm->x[i] = v;
    }
    if (tid < 32) {
        float mv = 0.f;
        if (tid == 0) mv = 1.f;
        else if (tid < S_NODE) mv = hmask[(size_t)n * LHID + tid - 1] ? 1.f : 0.f;
        sm->smask[tid] = mv;
    }
    __syncthreads();
    encoder(sm, S_NODE, &prm, tid);
    ovr[(size_t)n * EMBED + tid] = sm->x[tid];              // nenc[:,0]
}

// ---- small utility kernels ------------------------------------------------
__global__ void cvt_f32_f16_kernel(const float* __restrict__ s,
                                   _Float16* __restrict__ d, int n) {
    int i = blockIdx.x * 256 + threadIdx.x;
    if (i < n) d[i] = (_Float16)s[i];
}
__global__ void fill_inv_kernel(int* inv, int n) {
    int i = blockIdx.x * 256 + threadIdx.x;
    if (i < n) inv[i] = -1;
}
__global__ void scat_inv_kernel(int* __restrict__ inv, const int* __restrict__ subg, int n) {
    int i = blockIdx.x * 256 + threadIdx.x;
    if (i < n) inv[subg[i]] = i;
}
__global__ void gather_out_kernel(const float* __restrict__ hedge,
                                  const int* __restrict__ pe, float* __restrict__ out) {
    out[(size_t)blockIdx.x * EMBED + threadIdx.x] =
        hedge[(size_t)pe[blockIdx.x] * EMBED + threadIdx.x];
}

// ---------------------------------------------------------------------------
extern "C" void kernel_launch(void* const* d_in, const int* in_sizes, int n_in,
                              void* d_out, int out_size, void* d_ws, size_t ws_size,
                              hipStream_t stream) {
    const float* unity   = (const float*)d_in[0];
    const float* pos_tab = (const float*)d_in[1];
    const float* pad_emb = (const float*)d_in[2];
    const float* cls_emb = (const float*)d_in[3];
    const float* qkv_w   = (const float*)d_in[4];
    const float* qkv_b   = (const float*)d_in[5];
    const float* out_w   = (const float*)d_in[6];
    const float* out_b   = (const float*)d_in[7];
    const float* ln1_w   = (const float*)d_in[8];
    const float* ln1_b   = (const float*)d_in[9];
    const float* ln2_w   = (const float*)d_in[10];
    const float* ln2_b   = (const float*)d_in[11];
    const float* ff1_w   = (const float*)d_in[12];
    const float* ff1_b   = (const float*)d_in[13];
    const float* ff2_w   = (const float*)d_in[14];
    const float* ff2_b   = (const float*)d_in[15];
    const int*   xid     = (const int*)d_in[16];
    const unsigned char* xmask = (const unsigned char*)d_in[17];
    const int*   xpos    = (const int*)d_in[18];
    const int*   hid     = (const int*)d_in[19];
    const unsigned char* hmask = (const unsigned char*)d_in[20];
    const int*   subg    = (const int*)d_in[21];
    const int*   pe      = (const int*)d_in[22];

    // workspace layout
    const int nQKV = NLAYER * 3 * EMBED * EMBED;   // 393216
    const int nOUT = NLAYER * EMBED * EMBED;       // 131072
    const int nFF1 = NLAYER * HIDDEN * EMBED;      // 524288
    const int nFF2 = NLAYER * EMBED * HIDDEN;      // 524288
    size_t off = 0;
    auto take = [&](size_t bytes) { size_t o = off; off = (off + bytes + 255) & ~(size_t)255; return o; };
    char* ws = (char*)d_ws;
    _Float16* wqkv = (_Float16*)(ws + take((size_t)nQKV * 2));
    _Float16* wout = (_Float16*)(ws + take((size_t)nOUT * 2));
    _Float16* wff1 = (_Float16*)(ws + take((size_t)nFF1 * 2));
    _Float16* wff2 = (_Float16*)(ws + take((size_t)nFF2 * 2));
    float*    hedge = (float*)(ws + take((size_t)NEDGE * EMBED * 4));
    float*    ovr   = (float*)(ws + take((size_t)NNODE * EMBED * 4));
    int*      inv   = (int*)(ws + take((size_t)NVOC * 4));

    cvt_f32_f16_kernel<<<(nQKV + 255) / 256, 256, 0, stream>>>(qkv_w, wqkv, nQKV);
    cvt_f32_f16_kernel<<<(nOUT + 255) / 256, 256, 0, stream>>>(out_w, wout, nOUT);
    cvt_f32_f16_kernel<<<(nFF1 + 255) / 256, 256, 0, stream>>>(ff1_w, wff1, nFF1);
    cvt_f32_f16_kernel<<<(nFF2 + 255) / 256, 256, 0, stream>>>(ff2_w, wff2, nFF2);
    fill_inv_kernel<<<(NVOC + 255) / 256, 256, 0, stream>>>(inv, NVOC);
    scat_inv_kernel<<<(NNODE + 255) / 256, 256, 0, stream>>>(inv, subg, NNODE);

    EncParams prm;
    prm.wqkv = wqkv; prm.wout = wout; prm.wff1 = wff1; prm.wff2 = wff2;
    prm.bqkv = qkv_b; prm.bout = out_b; prm.bff1 = ff1_b; prm.bff2 = ff2_b;
    prm.ln1w = ln1_w; prm.ln1b = ln1_b; prm.ln2w = ln2_w; prm.ln2b = ln2_b;

    const size_t smem = sizeof(EncSM);
    (void)hipFuncSetAttribute((const void*)edge_encoder_kernel,
                              hipFuncAttributeMaxDynamicSharedMemorySize, (int)smem);
    (void)hipFuncSetAttribute((const void*)node_encoder_kernel,
                              hipFuncAttributeMaxDynamicSharedMemorySize, (int)smem);

    for (int k = 0; k <= 2; ++k) {
        edge_encoder_kernel<<<NEDGE, 256, smem, stream>>>(
            unity, pos_tab, xid, xmask, xpos, ovr, inv, (k > 0) ? 1 : 0, prm, hedge);
        if (k < 2) {
            node_encoder_kernel<<<NNODE, 256, smem, stream>>>(
                hedge, pad_emb, cls_emb, hid, hmask, prm, ovr);
        } else {
            gather_out_kernel<<<NBATCH, 256, 0, stream>>>(hedge, pe, (float*)d_out);
        }
    }
    (void)in_sizes; (void)n_in; (void)out_size; (void)ws_size;
}
